// SelfAttention_83631603188210
// MI455X (gfx1250) — compile-verified, matrix operand order checked by reading, not run on hardware
//
#include <hip/hip_runtime.h>

typedef unsigned short u16;
typedef __attribute__((ext_vector_type(16))) __bf16 v16bf;
typedef __attribute__((ext_vector_type(8)))  float  v8f;

union Frag { v16bf v; uint4 u[2]; };

__device__ __forceinline__ u16 f2bf(float f) {
    unsigned u = __builtin_bit_cast(unsigned, f);
    u += 0x7FFFu + ((u >> 16) & 1u);   // round-to-nearest-even
    return (u16)(u >> 16);
}

// ---------------- Tensor Data Mover (TDM) 2D tile load: global -> LDS ----------------
#if __has_builtin(__builtin_amdgcn_tensor_load_to_lds) && __has_builtin(__builtin_amdgcn_s_wait_tensorcnt)
#define USE_TDM 1
typedef __attribute__((ext_vector_type(4))) unsigned int u32x4;
typedef __attribute__((ext_vector_type(4))) int i32x4;
typedef __attribute__((ext_vector_type(8))) int i32x8;

// Loads a tile_h x tile_w (elements, 2 bytes each) tile whose rows are
// row_stride elements apart in global memory into packed LDS at lds_off.
// D# layout per CDNA5 ISA (group0 128b, group1 256b).
// 6-arg builtin form: (g0, g1, g2, g3, g4, cpol).
__device__ __forceinline__ void tdm_load_2d_bf16(const u16* gptr, unsigned lds_off,
                                                 unsigned tile_w, unsigned tile_h,
                                                 unsigned row_stride) {
    unsigned long long ga = (unsigned long long)gptr;
    u32x4 g0;
    g0[0] = 1u;                                   // count=1 (valid), user mode, no gather
    g0[1] = lds_off;                              // lds_addr (bytes)
    g0[2] = (unsigned)(ga & 0xFFFFFFFFu);         // global_addr[31:0]
    g0[3] = (unsigned)((ga >> 32) & 0x01FFFFFFu)  // global_addr[56:32]
          | (2u << 30);                           // type = 2 ("image")
    unsigned tw = row_stride;                     // tensor_dim0 (>= tile_w, no OOB)
    unsigned th = tile_h;                         // tensor_dim1
    i32x8 g1;
    g1[0] = (int)(1u << 16);                      // wg_mask=0, data_size=1 (2 bytes)
    g1[1] = (int)((tw & 0xFFFFu) << 16);          // tensor_dim0[15:0] at bits 63:48
    g1[2] = (int)((tw >> 16) | ((th & 0xFFFFu) << 16));
    g1[3] = (int)((th >> 16) | (tile_w << 16));   // tile_dim0 at bits 127:112
    g1[4] = (int)tile_h;                          // tile_dim1 (tile_dim2 = 0)
    g1[5] = (int)row_stride;                      // tensor_dim0_stride[31:0]
    g1[6] = 0;                                    // stride[47:32]=0, dim1_stride lo=0
    g1[7] = 0;
    i32x4 gz4 = {0, 0, 0, 0};
    i32x8 gz8 = {0, 0, 0, 0, 0, 0, 0, 0};
    __builtin_amdgcn_tensor_load_to_lds(g0, g1, gz4, gz4, gz8, 0);
}
__device__ __forceinline__ unsigned lds_offset_of(const void* p) {
    return (unsigned)(unsigned long long)p;       // generic shared addr low 32b = LDS offset
}
#else
#define USE_TDM 0
#endif

// ---------------- fp32 -> bf16 conversion ----------------
__global__ __launch_bounds__(256) void cvt_f32_bf16(const float* __restrict__ in,
                                                    u16* __restrict__ out, int n) {
    int i = (blockIdx.x * blockDim.x + threadIdx.x) * 2;
    if (i < n) {
        float2 f = *(const float2*)(in + i);
        unsigned p = (unsigned)f2bf(f.x) | ((unsigned)f2bf(f.y) << 16);
        *(unsigned*)(out + i) = p;
    }
}

// ---------------- bf16 GEMM + bias:  C[M,N] = A[M,K] * B[K,N] + bias ----------------
#define BM 128
#define BN 128
#define BK 32

template<bool OUT_BF16>
__global__ __launch_bounds__(256) void gemm_bf16(const u16* __restrict__ A,
                                                 const u16* __restrict__ B,
                                                 const float* __restrict__ bias,
                                                 void* __restrict__ C,
                                                 int M, int N, int K) {
    __shared__ u16 As[BM][BK];     // row-major A tile (filled by TDM when available)
    __shared__ u16 Bs[BN][BK];     // B tile stored transposed (col-contiguous)

    const int tid  = threadIdx.x;
    const int lane = tid & 31;
    const int wave = tid >> 5;             // 0..7, arranged 2x4
    const int wm   = (wave >> 2) * 64;     // wave M offset in block tile
    const int wn   = (wave & 3) * 32;      // wave N offset
    const int bm   = blockIdx.y * BM;
    const int bn   = blockIdx.x * BN;
    const int hs   = lane >> 4;            // half-select for 16-bit fragment layout
    const int l15  = lane & 15;

    const v8f vzero = {0.f,0.f,0.f,0.f,0.f,0.f,0.f,0.f};
    v8f acc[4][2];
    #pragma unroll
    for (int mt = 0; mt < 4; ++mt)
        #pragma unroll
        for (int nt = 0; nt < 2; ++nt) acc[mt][nt] = vzero;

    const int bk    = tid >> 3;            // 0..31
    const int bn16  = (tid & 7) * 16;      // 0..112
#if !USE_TDM
    const int arow  = tid >> 1;            // 0..127
    const int ahalf = (tid & 1) * 16;      // 0 / 16 elements
#endif

    for (int k0 = 0; k0 < K; k0 += BK) {
#if USE_TDM
        // A tile via Tensor Data Mover (wave-level DMA, tracked by TENSORcnt)
        if (wave == 0)
            tdm_load_2d_bf16(A + (size_t)bm * K + k0, lds_offset_of(&As[0][0]),
                             BK, BM, (unsigned)K);
#else
        const u16* ap = A + (size_t)(bm + arow) * K + k0 + ahalf;
        uint4 a0 = *(const uint4*)ap;
        uint4 a1 = *(const uint4*)(ap + 8);
        *(uint4*)&As[arow][ahalf]     = a0;
        *(uint4*)&As[arow][ahalf + 8] = a1;
#endif
        // stage B tile transposed into LDS
        const u16* bp = B + (size_t)(k0 + bk) * N + bn + bn16;
        u16 tmp[16];
        *(uint4*)&tmp[0] = *(const uint4*)bp;
        *(uint4*)&tmp[8] = *(const uint4*)(bp + 8);
        #pragma unroll
        for (int i = 0; i < 16; ++i) Bs[bn16 + i][bk] = tmp[i];

        if (k0 + BK < K)                   // prefetch next B k-tile (global_prefetch_b8)
            __builtin_prefetch(bp + (size_t)BK * N, 0, 1);

#if USE_TDM
        if (wave == 0) __builtin_amdgcn_s_wait_tensorcnt(0);
#endif
        __syncthreads();

        Frag af[4], bf[2];
        #pragma unroll
        for (int mt = 0; mt < 4; ++mt) {
            int r = wm + mt * 16 + l15;
            af[mt].u[0] = *(const uint4*)&As[r][hs * 8];
            af[mt].u[1] = *(const uint4*)&As[r][16 + hs * 8];
        }
        #pragma unroll
        for (int nt = 0; nt < 2; ++nt) {
            int c = wn + nt * 16 + l15;
            bf[nt].u[0] = *(const uint4*)&Bs[c][hs * 8];
            bf[nt].u[1] = *(const uint4*)&Bs[c][16 + hs * 8];
        }
        #pragma unroll
        for (int mt = 0; mt < 4; ++mt)
            #pragma unroll
            for (int nt = 0; nt < 2; ++nt)
                acc[mt][nt] = __builtin_amdgcn_wmma_f32_16x16x32_bf16(
                    false, af[mt].v, false, bf[nt].v, (short)0, acc[mt][nt], false, false);
        __syncthreads();
    }

    #pragma unroll
    for (int mt = 0; mt < 4; ++mt) {
        #pragma unroll
        for (int nt = 0; nt < 2; ++nt) {
            int col = bn + wn + nt * 16 + l15;
            float bv = bias[col];
            #pragma unroll
            for (int r = 0; r < 8; ++r) {
                int row = bm + wm + mt * 16 + ((lane < 16) ? r : r + 8);
                float val = acc[mt][nt][r] + bv;
                if (OUT_BF16)
                    ((u16*)C)[(size_t)row * N + col] = f2bf(val);
                else
                    ((float*)C)[(size_t)row * N + col] = val;
            }
        }
    }
}

// ---------------- fused flash-style attention, one (b,h,64-row q-tile) per block ----------------
__global__ __launch_bounds__(128) void attn_kernel(const u16* __restrict__ Q,
                                                   const u16* __restrict__ Kg,
                                                   const u16* __restrict__ Vg,
                                                   u16* __restrict__ O,
                                                   int S, int D) {
    __shared__ u16  Kt[64][64];      // K tile, rows = kv index, cols = dh (TDM-filled)
    __shared__ u16  Vt[64][64];      // V tile transposed: [dh][kv]
    __shared__ float Sb[64][64];     // scores (fp32)
    __shared__ u16  Pb[64][64];      // probabilities (bf16)
    __shared__ float mrow[64], lrow[64], frow[64];

    const int tid  = threadIdx.x;
    const int lane = tid & 31;
    const int wave = tid >> 5;           // 0..3, each owns a 16-row stripe
    const int hs   = lane >> 4;
    const int l15  = lane & 15;
    const int q0   = blockIdx.x * 64;
    const size_t head_off = (size_t)blockIdx.z * S * D + (size_t)blockIdx.y * 64;

    if (tid < 64) { mrow[tid] = -1e30f; lrow[tid] = 0.f; }

    // preload Q fragments for this wave's 16 rows (Dh = 64 -> two K=32 fragments)
    Frag qa0, qa1;
    {
        const u16* qp = Q + head_off + (size_t)(q0 + wave * 16 + l15) * D;
        qa0.u[0] = *(const uint4*)(qp + hs * 8);
        qa0.u[1] = *(const uint4*)(qp + 16 + hs * 8);
        qa1.u[0] = *(const uint4*)(qp + 32 + hs * 8);
        qa1.u[1] = *(const uint4*)(qp + 48 + hs * 8);
    }

    const v8f vzero = {0.f,0.f,0.f,0.f,0.f,0.f,0.f,0.f};
    v8f oacc[4];
    #pragma unroll
    for (int nt = 0; nt < 4; ++nt) oacc[nt] = vzero;

    const int lr = tid >> 1;             // row 0..63 for cooperative loads
    const int lh = (tid & 1) * 32;       // 0 / 32 element half

    for (int kv0 = 0; kv0 < S; kv0 += 64) {
        __syncthreads();                 // protect LDS reuse across iterations
#if USE_TDM
        // K tile via Tensor Data Mover while all threads transpose V manually
        if (wave == 0)
            tdm_load_2d_bf16(Kg + head_off + (size_t)kv0 * D,
                             lds_offset_of(&Kt[0][0]), 64, 64, (unsigned)D);
#else
        const u16* kp = Kg + head_off + (size_t)(kv0 + lr) * D + lh;
        *(uint4*)&Kt[lr][lh]      = *(const uint4*)(kp);
        *(uint4*)&Kt[lr][lh + 8]  = *(const uint4*)(kp + 8);
        *(uint4*)&Kt[lr][lh + 16] = *(const uint4*)(kp + 16);
        *(uint4*)&Kt[lr][lh + 24] = *(const uint4*)(kp + 24);
#endif
        // V tile, transposed into LDS so PV B-fragments are contiguous
        const u16* vp = Vg + head_off + (size_t)(kv0 + lr) * D + lh;
        u16 vt[32];
        *(uint4*)&vt[0]  = *(const uint4*)(vp);
        *(uint4*)&vt[8]  = *(const uint4*)(vp + 8);
        *(uint4*)&vt[16] = *(const uint4*)(vp + 16);
        *(uint4*)&vt[24] = *(const uint4*)(vp + 24);
        #pragma unroll
        for (int i = 0; i < 32; ++i) Vt[lh + i][lr] = vt[i];
#if USE_TDM
        if (wave == 0) __builtin_amdgcn_s_wait_tensorcnt(0);
#endif
        __syncthreads();

        // S = (Q K^T) / sqrt(Dh)
        #pragma unroll
        for (int nt = 0; nt < 4; ++nt) {
            int n = nt * 16 + l15;
            Frag kb0, kb1;
            kb0.u[0] = *(const uint4*)&Kt[n][hs * 8];
            kb0.u[1] = *(const uint4*)&Kt[n][16 + hs * 8];
            kb1.u[0] = *(const uint4*)&Kt[n][32 + hs * 8];
            kb1.u[1] = *(const uint4*)&Kt[n][48 + hs * 8];
            v8f s = vzero;
            s = __builtin_amdgcn_wmma_f32_16x16x32_bf16(false, qa0.v, false, kb0.v, (short)0, s, false, false);
            s = __builtin_amdgcn_wmma_f32_16x16x32_bf16(false, qa1.v, false, kb1.v, (short)0, s, false, false);
            #pragma unroll
            for (int r = 0; r < 8; ++r) {
                int row = wave * 16 + ((lane < 16) ? r : r + 8);
                Sb[row][nt * 16 + l15] = s[r] * 0.125f;
            }
        }
        __syncthreads();

        // online softmax: per-row running max / sum, P in bf16, rescale factor
        if (tid < 64) {
            int row = tid;
            float rm = -1e30f;
            for (int c = 0; c < 64; ++c) rm = fmaxf(rm, Sb[row][c]);
            float mo = mrow[row];
            float mn = fmaxf(mo, rm);
            float corr = __expf(mo - mn);
            float sum = 0.f;
            for (int c = 0; c < 64; ++c) {
                float p = __expf(Sb[row][c] - mn);
                Pb[row][c] = f2bf(p);
                sum += p;
            }
            mrow[row] = mn;
            lrow[row] = lrow[row] * corr + sum;
            frow[row] = corr;
        }
        __syncthreads();

        // rescale O accumulators
        #pragma unroll
        for (int nt = 0; nt < 4; ++nt)
            #pragma unroll
            for (int r = 0; r < 8; ++r) {
                int row = wave * 16 + ((lane < 16) ? r : r + 8);
                oacc[nt][r] *= frow[row];
            }

        // O += P * V
        Frag pa0, pa1;
        {
            int row = wave * 16 + l15;
            pa0.u[0] = *(const uint4*)&Pb[row][hs * 8];
            pa0.u[1] = *(const uint4*)&Pb[row][16 + hs * 8];
            pa1.u[0] = *(const uint4*)&Pb[row][32 + hs * 8];
            pa1.u[1] = *(const uint4*)&Pb[row][48 + hs * 8];
        }
        #pragma unroll
        for (int nt = 0; nt < 4; ++nt) {
            int dh = nt * 16 + l15;
            Frag vb0, vb1;
            vb0.u[0] = *(const uint4*)&Vt[dh][hs * 8];
            vb0.u[1] = *(const uint4*)&Vt[dh][16 + hs * 8];
            vb1.u[0] = *(const uint4*)&Vt[dh][32 + hs * 8];
            vb1.u[1] = *(const uint4*)&Vt[dh][48 + hs * 8];
            oacc[nt] = __builtin_amdgcn_wmma_f32_16x16x32_bf16(false, pa0.v, false, vb0.v, (short)0, oacc[nt], false, false);
            oacc[nt] = __builtin_amdgcn_wmma_f32_16x16x32_bf16(false, pa1.v, false, vb1.v, (short)0, oacc[nt], false, false);
        }
    }

    __syncthreads();
    // O /= l, store bf16
    #pragma unroll
    for (int nt = 0; nt < 4; ++nt) {
        int col = nt * 16 + l15;
        #pragma unroll
        for (int r = 0; r < 8; ++r) {
            int rit = wave * 16 + ((lane < 16) ? r : r + 8);
            float inv = 1.0f / lrow[rit];
            O[head_off + (size_t)(q0 + rit) * D + col] = f2bf(oacc[nt][r] * inv);
        }
    }
}

extern "C" void kernel_launch(void* const* d_in, const int* in_sizes, int n_in,
                              void* d_out, int out_size, void* d_ws, size_t ws_size,
                              hipStream_t stream) {
    const float* x  = (const float*)d_in[0];
    const float* bq = (const float*)d_in[2];
    const float* bk = (const float*)d_in[4];
    const float* bv = (const float*)d_in[6];
    const float* bo = (const float*)d_in[8];
    float* out = (float*)d_out;

    const int Bb = 4, S = 2048, D = 1024, H = 16;
    const size_t MS = (size_t)Bb * S;               // 8192 rows
    const size_t actN = MS * D;                     // activation elements
    const size_t wN   = (size_t)D * D;              // weight elements

    char* ws = (char*)d_ws;
    u16* xb  = (u16*)ws;  ws += actN * 2;
    u16* wqb = (u16*)ws;  ws += wN * 2;
    u16* wkb = (u16*)ws;  ws += wN * 2;
    u16* wvb = (u16*)ws;  ws += wN * 2;
    u16* wob = (u16*)ws;  ws += wN * 2;
    u16* Qb  = (u16*)ws;  ws += actN * 2;
    u16* Kb  = (u16*)ws;  ws += actN * 2;
    u16* Vb  = (u16*)ws;  ws += actN * 2;
    u16* Ab  = (u16*)ws;  ws += actN * 2;

    // fp32 -> bf16 conversions
    cvt_f32_bf16<<<(unsigned)((actN / 2 + 255) / 256), 256, 0, stream>>>(x, xb, (int)actN);
    cvt_f32_bf16<<<(unsigned)((wN / 2 + 255) / 256), 256, 0, stream>>>((const float*)d_in[1], wqb, (int)wN);
    cvt_f32_bf16<<<(unsigned)((wN / 2 + 255) / 256), 256, 0, stream>>>((const float*)d_in[3], wkb, (int)wN);
    cvt_f32_bf16<<<(unsigned)((wN / 2 + 255) / 256), 256, 0, stream>>>((const float*)d_in[5], wvb, (int)wN);
    cvt_f32_bf16<<<(unsigned)((wN / 2 + 255) / 256), 256, 0, stream>>>((const float*)d_in[7], wob, (int)wN);

    // Q/K/V projections (bf16 out, bias fused)
    dim3 gg(D / BN, (unsigned)(MS / BM));
    gemm_bf16<true><<<gg, 256, 0, stream>>>(xb, wqb, bq, Qb, (int)MS, D, D);
    gemm_bf16<true><<<gg, 256, 0, stream>>>(xb, wkb, bk, Kb, (int)MS, D, D);
    gemm_bf16<true><<<gg, 256, 0, stream>>>(xb, wvb, bv, Vb, (int)MS, D, D);

    // fused attention
    dim3 ga(S / 64, H, Bb);
    attn_kernel<<<ga, 128, 0, stream>>>(Qb, Kb, Vb, Ab, S, D);

    // output projection (fp32 out, bias fused)
    gemm_bf16<false><<<gg, 256, 0, stream>>>(Ab, wob, bo, out, (int)MS, D, D);
}
